// GatedSSMModel_62302795596638
// MI455X (gfx1250) — matrix operand order repaired and checked by reading, not compiled, not run on hardware
//
#include <hip/hip_runtime.h>

// Problem constants (from the reference)
#define B_     8
#define S_     512
#define D_     1024          // == H_*DH_
#define H_     16
#define DH_    64
#define N_     32
#define L_     2
#define VOCAB_ 128
#define M_     (B_*S_)       // 4096 rows of activations

typedef __attribute__((ext_vector_type(16))) _Float16 v16h;
typedef __attribute__((ext_vector_type(8)))  _Float16 v8h;
typedef __attribute__((ext_vector_type(8)))  float    v8f;

// ---------------------------------------------------------------------------
// Embedding + positional add:  x[b,s,:] = emb[tok[b,s],:] + pos[s,:]
// grid = B*S blocks, 256 threads, 4 f32 each
// ---------------------------------------------------------------------------
__global__ __launch_bounds__(256) void embed_kernel(
    const int* __restrict__ tokens, const float* __restrict__ emb,
    const float* __restrict__ pos, float* __restrict__ x)
{
    const int row = blockIdx.x;            // b*S + s
    const int s   = row & (S_ - 1);
    const int tok = tokens[row];
    const float* e = emb + (size_t)tok * D_;
    const float* p = pos + (size_t)s * D_;
    float* xr = x + (size_t)row * D_;
#pragma unroll
    for (int i = 0; i < 4; ++i) {
        const int d = threadIdx.x + 256 * i;
        xr[d] = e[d] + p[d];
    }
}

// ---------------------------------------------------------------------------
// LayerNorm over D=1024, writes f16 (WMMA A-operand is plain row-major f16)
// grid = rows, 256 threads, 4 elems/thread
// ---------------------------------------------------------------------------
__global__ __launch_bounds__(256) void ln_kernel(
    const float* __restrict__ x, const float* __restrict__ g,
    const float* __restrict__ b, _Float16* __restrict__ outh)
{
    const int row = blockIdx.x;
    const int t   = threadIdx.x;
    const float* xr = x + (size_t)row * D_;
    float v[4];
    float s = 0.f, s2 = 0.f;
#pragma unroll
    for (int i = 0; i < 4; ++i) {
        v[i] = xr[t + 256 * i];
        s  += v[i];
        s2 += v[i] * v[i];
    }
    __shared__ float ssum[256], ssq[256];
    ssum[t] = s; ssq[t] = s2;
    __syncthreads();
    for (int st = 128; st > 0; st >>= 1) {
        if (t < st) { ssum[t] += ssum[t + st]; ssq[t] += ssq[t + st]; }
        __syncthreads();
    }
    const float mean = ssum[0] * (1.f / D_);
    const float var  = ssq[0] * (1.f / D_) - mean * mean;   // biased var (jnp.var)
    const float inv  = rsqrtf(var + 1e-5f);
    _Float16* o = outh + (size_t)row * D_;
#pragma unroll
    for (int i = 0; i < 4; ++i) {
        const int d = t + 256 * i;
        o[d] = (_Float16)((v[i] - mean) * inv * g[d] + b[d]);
    }
}

// ---------------------------------------------------------------------------
// f32 -> f16 weight conversion
// ---------------------------------------------------------------------------
__global__ __launch_bounds__(256) void f32_to_f16_kernel(
    const float* __restrict__ src, _Float16* __restrict__ dst, int n)
{
    const int i = blockIdx.x * 256 + threadIdx.x;
    if (i < n) dst[i] = (_Float16)src[i];
}

// ---------------------------------------------------------------------------
// WMMA GEMM:  C[m,e] = sum_k A[m,k]*W[e,k] + bias[e]  (+ resid[m,e])
//   A: MxK row-major f16, W: ExK row-major f16 (einsum 'bsd,ed->bse')
//   f32 accumulate via v_wmma_f32_16x16x32_f16.
// Block = 64 threads (2 waves). Each wave owns a 32x64 output tile:
//   2 A fragments x 4 B fragments -> 8 WMMAs per K-step; each B fragment is
//   reused by both A fragments (halves L2 weight traffic vs 16x64 tiles,
//   ~22 FLOP/byte) and 8 independent WMMAs/step hide the XDL latency.
// Fragment layouts used here (contiguous k = 16h + s, same order in A and B):
//   A (16x32 f16): lane<16 -> row M=lane, halves = K{0..15};
//                  lane>=16 -> row M=lane-16, halves = K{16..31}
//   B (32x16 f16): col N=lane&15; lane<16 holds K{0..15}, lane>=16 holds
//                  K{16..31}; 16 contiguous halves (32B aligned).
//   C (16x16 f32): VGPR r -> row r (lanes 0-15) / row 8+r (lanes 16-31),
//                  col = lane&15.
// ---------------------------------------------------------------------------
template <bool RESID>
__global__ __launch_bounds__(64) void wmma_gemm_kernel(
    const _Float16* __restrict__ A,     // M x K
    const _Float16* __restrict__ W,     // E x K
    const float*    __restrict__ bias,  // E
    const float*    __restrict__ resid, // M x E (may alias C; same-elem RMW)
    float*          __restrict__ C,     // M x E
    int M, int K, int E)
{
    const int wave = threadIdx.x >> 5;
    const int lane = threadIdx.x & 31;
    const int m0 = blockIdx.x << 5;                 // 32-row tile per wave
    const int e0 = ((blockIdx.y << 1) + wave) << 6; // 64-col strip per wave

    const int arow = lane & 15;
    const int ak   = (lane < 16) ? 0 : 16;
    const int bcol = lane & 15;
    const int bk   = (lane < 16) ? 0 : 16;

    const _Float16* Ap0 = A + (size_t)(m0 + arow) * K + ak;
    const _Float16* Ap1 = Ap0 + (size_t)16 * K;
    const _Float16* Wp  = W + (size_t)(e0 + bcol) * K + bk;

    v8f acc[2][4];
#pragma unroll
    for (int mm = 0; mm < 2; ++mm)
#pragma unroll
        for (int n = 0; n < 4; ++n) acc[mm][n] = (v8f){};

    for (int k0 = 0; k0 < K; k0 += 32) {
        const v8h lo0 = *(const v8h*)(Ap0 + k0);       // K + {0..7}/{16..23}
        const v8h hi0 = *(const v8h*)(Ap0 + k0 + 8);   // K + {8..15}/{24..31}
        const v8h lo1 = *(const v8h*)(Ap1 + k0);
        const v8h hi1 = *(const v8h*)(Ap1 + k0 + 8);
        v16h a0, a1;
#pragma unroll
        for (int i = 0; i < 8; ++i) {
            a0[i] = lo0[i]; a0[i + 8] = hi0[i];
            a1[i] = lo1[i]; a1[i + 8] = hi1[i];
        }
#pragma unroll
        for (int n = 0; n < 4; ++n) {
            const v16h bfrag = *(const v16h*)(Wp + k0 + (size_t)(16 * n) * K);
            acc[0][n] = __builtin_amdgcn_wmma_f32_16x16x32_f16(
                false, a0, false, bfrag, (short)0, acc[0][n], false, false);
            acc[1][n] = __builtin_amdgcn_wmma_f32_16x16x32_f16(
                false, a1, false, bfrag, (short)0, acc[1][n], false, false);
        }
    }

    const int ccol  = lane & 15;
    const int rbase = (lane < 16) ? 0 : 8;
#pragma unroll
    for (int mm = 0; mm < 2; ++mm) {
#pragma unroll
        for (int n = 0; n < 4; ++n) {
            const int e = e0 + n * 16 + ccol;
            const float bv = bias[e];
#pragma unroll
            for (int r = 0; r < 8; ++r) {
                const size_t idx = (size_t)(m0 + mm * 16 + rbase + r) * E + e;
                float val = acc[mm][n][r] + bv;
                if (RESID) val += resid[idx];
                C[idx] = val;
            }
        }
    }
}

// ---------------------------------------------------------------------------
// Gated SSM scan.  One block per (b,h) pair; state h is 32x64 f32 held in
// registers: thread t owns j = t&63, i-range (t>>6)*8 .. +7 (8 elems).
// Per step: h[i][j] = alpha[i]*h[i][j] + k[i]*v[j];  o[j] = sum_i q[i]*h[i][j]
// Output: og[b,t,h,j] = o[j] * silu(g[b,t,h,j])  (f16, feeds Wo GEMM)
// ---------------------------------------------------------------------------
__global__ __launch_bounds__(256) void ssm_scan_kernel(
    const float* __restrict__ Q, const float* __restrict__ K,
    const float* __restrict__ V, const float* __restrict__ G,
    const float* __restrict__ alog,   // H*N (this layer)
    _Float16* __restrict__ OgH)
{
    const int b  = blockIdx.x / H_;
    const int h  = blockIdx.x % H_;
    const int t  = threadIdx.x;
    const int j  = t & (DH_ - 1);
    const int ig = t >> 6;            // 0..3

    float hreg[8];
    float alpha[8];
#pragma unroll
    for (int ii = 0; ii < 8; ++ii) {
        hreg[ii] = 0.f;
        const float al = alog[h * N_ + ig * 8 + ii];
        alpha[ii] = 1.f / (1.f + __expf(-al));      // sigmoid
    }

    __shared__ float sk[N_], sq[N_], sv[DH_], sg[DH_];
    __shared__ float red[4][DH_];

    const size_t base = (size_t)b * S_ * D_ + (size_t)h * DH_;

    for (int ts = 0; ts < S_; ++ts) {
        const size_t off = base + (size_t)ts * D_;
        if (t < DH_)                { sv[t] = V[off + t]; sg[t] = G[off + t]; }
        else if (t < DH_ + N_)      { sk[t - DH_] = K[off + (t - DH_)]; }
        else if (t < DH_ + 2 * N_)  { sq[t - DH_ - N_] = Q[off + (t - DH_ - N_)]; }
        __syncthreads();

        const float vj = sv[j];
        float osum = 0.f;
#pragma unroll
        for (int ii = 0; ii < 8; ++ii) {
            const int i = ig * 8 + ii;
            hreg[ii] = alpha[ii] * hreg[ii] + sk[i] * vj;
            osum += sq[i] * hreg[ii];
        }
        red[ig][j] = osum;
        __syncthreads();

        if (ig == 0) {
            const float o  = red[0][j] + red[1][j] + red[2][j] + red[3][j];
            const float gv = sg[j];
            const float gate = gv / (1.f + __expf(-gv));   // silu
            OgH[off + j] = (_Float16)(o * gate);
        }
        __syncthreads();   // protect shared buffers before next step's fill
    }
}

// ---------------------------------------------------------------------------
// Launch: embed -> L x [LN, 5 wcvt, 4 WMMA GEMMs, scan, WMMA GEMM(+resid)]
//         -> final LN -> head WMMA GEMM into d_out.
// Workspace layout (~107 MB, MB-aligned so all vector loads stay aligned).
// ---------------------------------------------------------------------------
extern "C" void kernel_launch(void* const* d_in, const int* in_sizes, int n_in,
                              void* d_out, int out_size, void* d_ws, size_t ws_size,
                              hipStream_t stream)
{
    (void)in_sizes; (void)n_in; (void)out_size; (void)ws_size;

    const int*   tokens = (const int*)  d_in[0];
    const float* emb    = (const float*)d_in[1];
    const float* pos    = (const float*)d_in[2];
    const float* ln_g   = (const float*)d_in[3];
    const float* ln_b   = (const float*)d_in[4];
    const float* wq     = (const float*)d_in[5];
    const float* bq     = (const float*)d_in[6];
    const float* wk     = (const float*)d_in[7];
    const float* bk     = (const float*)d_in[8];
    const float* wv     = (const float*)d_in[9];
    const float* bv     = (const float*)d_in[10];
    const float* wg     = (const float*)d_in[11];
    const float* bg     = (const float*)d_in[12];
    const float* wo     = (const float*)d_in[13];
    const float* bo     = (const float*)d_in[14];
    const float* alog   = (const float*)d_in[15];
    const float* fn_g   = (const float*)d_in[16];
    const float* fn_b   = (const float*)d_in[17];
    const float* head_w = (const float*)d_in[18];
    const float* head_b = (const float*)d_in[19];

    char* ws = (char*)d_ws;
    const size_t MB = 1024 * 1024;
    float*    x    = (float*)   (ws + 0);         // 16 MB  activations (f32)
    _Float16* xnh  = (_Float16*)(ws + 16 * MB);   //  8 MB  LN output (f16)
    float*    qf   = (float*)   (ws + 24 * MB);   // 16 MB
    float*    kf   = (float*)   (ws + 40 * MB);   // 16 MB
    float*    vf   = (float*)   (ws + 56 * MB);   // 16 MB
    float*    gf   = (float*)   (ws + 72 * MB);   // 16 MB
    _Float16* ogh  = (_Float16*)(ws + 88 * MB);   //  8 MB  gated output (f16)
    _Float16* wh[5];
    for (int i = 0; i < 5; ++i)
        wh[i] = (_Float16*)(ws + 96 * MB + (size_t)i * 2 * MB);  // 10 MB
    _Float16* hwh  = (_Float16*)(ws + 106 * MB);  // 0.25 MB head weights (f16)

    embed_kernel<<<M_, 256, 0, stream>>>(tokens, emb, pos, x);

    const dim3 gemm_grid(M_ / 32, D_ / 128);        // 128 x 8 blocks, 2 waves ea
    const dim3 head_grid(M_ / 32, VOCAB_ / 128);    // 128 x 1

    for (int l = 0; l < L_; ++l) {
        ln_kernel<<<M_, 256, 0, stream>>>(x, ln_g + (size_t)l * D_,
                                          ln_b + (size_t)l * D_, xnh);

        const float* Wsrc[5] = {
            wq + (size_t)l * D_ * D_, wk + (size_t)l * D_ * D_,
            wv + (size_t)l * D_ * D_, wg + (size_t)l * D_ * D_,
            wo + (size_t)l * D_ * D_ };
        for (int i = 0; i < 5; ++i)
            f32_to_f16_kernel<<<(D_ * D_) / 256, 256, 0, stream>>>(Wsrc[i], wh[i],
                                                                   D_ * D_);

        wmma_gemm_kernel<false><<<gemm_grid, 64, 0, stream>>>(
            xnh, wh[0], bq + (size_t)l * D_, nullptr, qf, M_, D_, D_);
        wmma_gemm_kernel<false><<<gemm_grid, 64, 0, stream>>>(
            xnh, wh[1], bk + (size_t)l * D_, nullptr, kf, M_, D_, D_);
        wmma_gemm_kernel<false><<<gemm_grid, 64, 0, stream>>>(
            xnh, wh[2], bv + (size_t)l * D_, nullptr, vf, M_, D_, D_);
        wmma_gemm_kernel<false><<<gemm_grid, 64, 0, stream>>>(
            xnh, wh[3], bg + (size_t)l * D_, nullptr, gf, M_, D_, D_);

        ssm_scan_kernel<<<B_ * H_, 256, 0, stream>>>(
            qf, kf, vf, gf, alog + (size_t)l * H_ * N_, ogh);

        // y = og @ Wo^T + bo ; x = x + y   (fused residual, in-place per-elem)
        wmma_gemm_kernel<true><<<gemm_grid, 64, 0, stream>>>(
            ogh, wh[4], bo + (size_t)l * D_, x, x, M_, D_, D_);
    }

    ln_kernel<<<M_, 256, 0, stream>>>(x, fn_g, fn_b, xnh);
    f32_to_f16_kernel<<<(VOCAB_ * D_) / 256, 256, 0, stream>>>(head_w, hwh,
                                                               VOCAB_ * D_);
    wmma_gemm_kernel<false><<<head_grid, 64, 0, stream>>>(
        xnh, hwh, head_b, nullptr, (float*)d_out, M_, D_, VOCAB_);
}
